// TwicingCausalAttention_56160992362651
// MI455X (gfx1250) — matrix operand order repaired and checked by reading, not compiled, not run on hardware
//
#include <hip/hip_runtime.h>
#include <hip/hip_bf16.h>

typedef __attribute__((ext_vector_type(16))) _Float16 v16h;
typedef __attribute__((ext_vector_type(8)))  _Float16 v8h;
typedef __attribute__((ext_vector_type(8)))  float    v8f;
typedef __attribute__((ext_vector_type(4)))  unsigned int u32x4;
typedef __attribute__((ext_vector_type(8)))  int          i32x8;
typedef __attribute__((ext_vector_type(4)))  int          i32x4;

#define D_MODEL 1024
#define N_HEADS 16
#define D_HEAD  64
#define BATCH   2
#define SEQ     2048
#define M_ROWS  (BATCH * SEQ)          // 4096
#define QKV_N   (3 * D_MODEL)          // 3072

// ---------------------------------------------------------------------------
// fp32 -> fp16 conversion (all heavy math runs through WMMA in f16)
// ---------------------------------------------------------------------------
__global__ void cvt_f32_to_f16(const float* __restrict__ in,
                               _Float16* __restrict__ out, int n) {
    int i = blockIdx.x * blockDim.x + threadIdx.x;
    if (i < n) out[i] = (_Float16)in[i];
}

// ---------------------------------------------------------------------------
// WMMA fragment loaders for C = A (MxK, row-major) * B^T (B stored NxK row-major)
//
// A-fragment (16x32 f16): lane L<16 -> row L, K = 0..15;
// lane L>=16 -> row L-16, K = 16..31.  Two 16B contiguous loads.
// ---------------------------------------------------------------------------
__device__ __forceinline__ v16h load_a_frag(const _Float16* __restrict__ A,
                                            int lda, int rowBase, int k0, int lane) {
    int row = rowBase + (lane & 15);
    int off = (lane >> 4) * 16;
    v8h lo = *(const v8h*)(A + (size_t)row * lda + k0 + off);
    v8h hi = *(const v8h*)(A + (size_t)row * lda + k0 + 8 + off);
    v16h a;
#pragma unroll
    for (int i = 0; i < 8; ++i) { a[i] = lo[i]; a[8 + i] = hi[i]; }
    return a;
}

// B-fragment (32x16 f16): lane L<16 -> col L, contraction K = 0..15;
// lane L>=16 -> col L-16, K = 16..31.  Since B is stored NxK row-major,
// this is one contiguous 32B load per lane.
__device__ __forceinline__ v16h load_bt_frag(const _Float16* __restrict__ Bm,
                                             int ldb, int colBase, int k0, int lane) {
    int n = colBase + (lane & 15);
    int off = (lane >> 4) * 16;
    return *(const v16h*)(Bm + (size_t)n * ldb + k0 + off);
}

// ---------------------------------------------------------------------------
// Tensor Data Mover: DMA a rows x 64 f16 tile (contiguous, stride 64) into LDS.
// Descriptor per CDNA5 ISA; tensor == tile so no OOB logic.
// All argument values must be wave-uniform (they come from blockIdx / loop idx).
// ---------------------------------------------------------------------------
__device__ __forceinline__ void tdm_load_v_tile(unsigned lds_off,
                                                const _Float16* __restrict__ gptr) {
#if __has_builtin(__builtin_amdgcn_tensor_load_to_lds)
    unsigned long long ga = (unsigned long long)(size_t)gptr;
    u32x4 g0;
    g0[0] = 1u;                                   // count=1, user D#, no gather
    g0[1] = lds_off;                              // lds_addr (bytes)
    g0[2] = (unsigned)(ga & 0xFFFFFFFFu);         // global_addr[31:0]
    g0[3] = (unsigned)((ga >> 32) & 0x1FFFFFFu)   // global_addr[56:32]
          | (2u << 30);                           // type = 2 ("image")
    i32x8 g1;
    g1[0] = 1 << 16;                              // wg_mask=0, data_size=1 (2B)
    g1[1] = 64 << 16;                             // tensor_dim0 = 64
    g1[2] = 32 << 16;                             // tensor_dim1 = 32
    g1[3] = 64 << 16;                             // tile_dim0 = 64
    g1[4] = 32;                                   // tile_dim1 = 32, tile_dim2 = 0
    g1[5] = 64;                                   // tensor_dim0_stride = 64
    g1[6] = 0;
    g1[7] = 0;
    i32x4 g2 = {0, 0, 0, 0};                      // dims 2/3 unused
    i32x4 g3 = {0, 0, 0, 0};
#if defined(__clang_major__) && (__clang_major__ >= 23)
    i32x8 g4 = {0, 0, 0, 0, 0, 0, 0, 0};
    __builtin_amdgcn_tensor_load_to_lds(g0, g1, g2, g3, g4, 0);
#else
    __builtin_amdgcn_tensor_load_to_lds(g0, g1, g2, g3, 0);
#endif
    __builtin_amdgcn_s_wait_tensorcnt(0);
#else
    // Fallback: cooperative wave copy (lane L -> row L, 128B contiguous)
    int lane = threadIdx.x & 31;
    _Float16* dst = (_Float16*)((char*)0 + 0);    // unused; fallback below
    (void)dst;
    v8h* lrow = (v8h*)((char*)nullptr);
    (void)lrow; (void)lds_off; (void)gptr;
#endif
}

// ---------------------------------------------------------------------------
// QKV projection: QKV[m,n] = sum_k X[m,k] * Wqkv[n,k], scattered directly
// into per-head Q/K/V buffers laid out [B, H, T, D_HEAD] (f16).
// One wave -> 16x64 tile of C; block = 8 waves -> 128 rows x 64 cols.
// ---------------------------------------------------------------------------
__global__ __launch_bounds__(256)
void gemm_qkv(const _Float16* __restrict__ X, const _Float16* __restrict__ W,
              _Float16* __restrict__ Q, _Float16* __restrict__ K,
              _Float16* __restrict__ V) {
    int lane = threadIdx.x & 31;
    int wave = threadIdx.x >> 5;
    int rowBase = blockIdx.x * 128 + wave * 16;
    int colBase = blockIdx.y * 64;

    v8f acc[4] = {};
    for (int k0 = 0; k0 < D_MODEL; k0 += 32) {
        v16h a = load_a_frag(X, D_MODEL, rowBase, k0, lane);
        // pull the k-tile after next into cache (global_prefetch_b8)
        __builtin_prefetch(X + (size_t)(rowBase + (lane & 15)) * D_MODEL + k0 + 64, 0, 1);
#pragma unroll
        for (int nb = 0; nb < 4; ++nb) {
            v16h b = load_bt_frag(W, D_MODEL, colBase + nb * 16, k0, lane);
            acc[nb] = __builtin_amdgcn_wmma_f32_16x16x32_f16(
                false, a, false, b, (short)0, acc[nb], false, false);
        }
    }
    // C/D layout: VGPR r -> row r (lanes 0-15) or r+8 (lanes 16-31), col = lane&15
    int rowOff = (lane >> 4) * 8;
#pragma unroll
    for (int nb = 0; nb < 4; ++nb) {
        int n     = colBase + nb * 16 + (lane & 15);
        int which = n >> 10;          // 0=Q, 1=K, 2=V
        int head  = (n >> 6) & (N_HEADS - 1);
        int d     = n & (D_HEAD - 1);
        _Float16* dst = (which == 0) ? Q : (which == 1) ? K : V;
#pragma unroll
        for (int r = 0; r < 8; ++r) {
            int m = rowBase + r + rowOff;
            int b = m >> 11;          // SEQ = 2048
            int t = m & (SEQ - 1);
            dst[(((size_t)b * N_HEADS + head) * SEQ + t) * D_HEAD + d] =
                (_Float16)acc[nb][r];
        }
    }
}

// ---------------------------------------------------------------------------
// Causal flash attention. Block = 4 waves, each owning one 16-query tile of a
// shared 64-query super-tile. All waves walk the same kv blocks so that the
// 32x64 V tile can be staged once per block into LDS by the Tensor Data Mover
// (wave 3 issues tensor_load_to_lds + s_wait_tensorcnt; barriers publish it).
// Used twice: O = softmax_causal(Q K^T / 8) @ Vsrc   (Vsrc = V, then Vsrc = AV)
// ---------------------------------------------------------------------------
__global__ __launch_bounds__(128)
void flash_attn(const _Float16* __restrict__ Q, const _Float16* __restrict__ K,
                const _Float16* __restrict__ V, _Float16* __restrict__ O) {
    __shared__ __align__(16) _Float16 ptile[4][16][32];   // per-wave P staging
    __shared__ __align__(16) _Float16 vtile[32][64];      // shared V tile (TDM dest)

    int lane = threadIdx.x & 31;
    int wave = threadIdx.x >> 5;
    int half = lane >> 4;             // which 16-lane half
    int lcol = lane & 15;
    int rowOff = half * 8;
    int qbase  = blockIdx.x * 64 + wave * 16;
    int bh     = blockIdx.y;

    const _Float16* Qb = Q + (size_t)bh * SEQ * D_HEAD;
    const _Float16* Kb = K + (size_t)bh * SEQ * D_HEAD;
    const _Float16* Vb = V + (size_t)bh * SEQ * D_HEAD;
    _Float16*       Ob = O + (size_t)bh * SEQ * D_HEAD;

    const float scale = 0.125f;       // 1/sqrt(64)

    // Q tile as two A-fragments (d = 0..31, 32..63)
    v16h qa[2];
#pragma unroll
    for (int kc = 0; kc < 2; ++kc)
        qa[kc] = load_a_frag(Qb, D_HEAD, qbase, kc * 32, lane);

    v8f  o[4] = {};
    float mrow[8], lrow[8];
#pragma unroll
    for (int r = 0; r < 8; ++r) { mrow[r] = -__builtin_inff(); lrow[r] = 0.f; }

    unsigned vtile_lds = (unsigned)(size_t)&vtile[0][0];
    int kvEnd = blockIdx.x * 64 + 64;          // covers wave 3's causal range

    for (int kv0 = 0; kv0 < kvEnd; kv0 += 32) {
        __syncthreads();                       // vtile consumers from prev iter done
        if (wave == 3) {                       // wave 3 is active on every iteration
#if __has_builtin(__builtin_amdgcn_tensor_load_to_lds)
            tdm_load_v_tile(vtile_lds, Vb + (size_t)kv0 * D_HEAD);
#else
            // cooperative copy: lane L -> V row kv0+L (128B contiguous)
            const v8h* src = (const v8h*)(Vb + (size_t)(kv0 + lane) * D_HEAD);
            v8h* dst = (v8h*)&vtile[lane][0];
#pragma unroll
            for (int j = 0; j < 8; ++j) dst[j] = src[j];
#endif
        }
        __syncthreads();                       // vtile published

        if (kv0 < qbase + 16) {                // this wave's causal range
            // --- S = Q * K^T for this 16x32 tile: 4 WMMAs -------------------
            v8f s[2] = {};
#pragma unroll
            for (int nblk = 0; nblk < 2; ++nblk)
#pragma unroll
                for (int kc = 0; kc < 2; ++kc) {
                    v16h kb = load_bt_frag(Kb, D_HEAD, kv0 + nblk * 16, kc * 32, lane);
                    s[nblk] = __builtin_amdgcn_wmma_f32_16x16x32_f16(
                        false, qa[kc], false, kb, (short)0, s[nblk], false, false);
                }

            // --- online softmax (rows live in VGPR index r) -----------------
#pragma unroll
            for (int r = 0; r < 8; ++r) {
                int row = qbase + r + rowOff;
                float s0 = s[0][r] * scale;
                float s1 = s[1][r] * scale;
                if (kv0 + lcol      > row) s0 = -1e30f;   // causal mask
                if (kv0 + 16 + lcol > row) s1 = -1e30f;

                float rmax = fmaxf(s0, s1);
#pragma unroll
                for (int mset = 1; mset < 16; mset <<= 1)  // stays in lane-half
                    rmax = fmaxf(rmax, __shfl_xor(rmax, mset, 32));

                float mnew  = fmaxf(mrow[r], rmax);
                float alpha = __expf(mrow[r] - mnew);
                float p0 = __expf(s0 - mnew);
                float p1 = __expf(s1 - mnew);
                float psum = p0 + p1;
#pragma unroll
                for (int mset = 1; mset < 16; mset <<= 1)
                    psum += __shfl_xor(psum, mset, 32);

                lrow[r] = lrow[r] * alpha + psum;
                mrow[r] = mnew;
#pragma unroll
                for (int nb = 0; nb < 4; ++nb) o[nb][r] *= alpha;

                // stage P (C-layout -> LDS row-major) for A-fragment reload
                ptile[wave][r + rowOff][lcol]      = (_Float16)p0;
                ptile[wave][r + rowOff][16 + lcol] = (_Float16)p1;
            }

            // --- reload P in A-fragment layout (16 rows x 32 keys) ----------
            v16h pa;
            {
                int off = half * 16;
                v8h lo = *(const v8h*)&ptile[wave][lcol][off];
                v8h hi = *(const v8h*)&ptile[wave][lcol][8 + off];
#pragma unroll
                for (int i = 0; i < 8; ++i) { pa[i] = lo[i]; pa[8 + i] = hi[i]; }
            }

            // --- O += P * V : 4 WMMAs, V B-fragments gathered from LDS ------
#pragma unroll
            for (int nb = 0; nb < 4; ++nb) {
                v16h vb;
                int d = nb * 16 + lcol;
#pragma unroll
                for (int j = 0; j < 16; ++j)
                    vb[j] = vtile[half * 16 + j][d];
                o[nb] = __builtin_amdgcn_wmma_f32_16x16x32_f16(
                    false, pa, false, vb, (short)0, o[nb], false, false);
            }
        }
    }

    // --- normalize and store (f16, per-head layout) ------------------------
#pragma unroll
    for (int r = 0; r < 8; ++r) {
        float inv = 1.f / lrow[r];
        int row = qbase + r + rowOff;
#pragma unroll
        for (int nb = 0; nb < 4; ++nb)
            Ob[(size_t)row * D_HEAD + nb * 16 + lcol] =
                (_Float16)(o[nb][r] * inv);
    }
}

// ---------------------------------------------------------------------------
// M[b,t,h*64+d] = 2*AV - AAV   ([B,H,T,D] -> [B*T, D_MODEL] f16)
// ---------------------------------------------------------------------------
__global__ void combine_av(const _Float16* __restrict__ AV,
                           const _Float16* __restrict__ AAV,
                           _Float16* __restrict__ Mh) {
    int i = blockIdx.x * blockDim.x + threadIdx.x;
    if (i >= BATCH * SEQ * D_MODEL) return;
    int d = i & (D_HEAD - 1);
    int h = (i >> 6) & (N_HEADS - 1);
    int t = (i >> 10) & (SEQ - 1);
    int b = i >> 21;                       // SEQ*D_MODEL = 2^21
    size_t src = (((size_t)b * N_HEADS + h) * SEQ + t) * D_HEAD + d;
    float av = (float)AV[src], aav = (float)AAV[src];
    Mh[i] = (_Float16)(2.f * av - aav);
}

// ---------------------------------------------------------------------------
// Output projection: out[m,n] = sum_k M[m,k] * Wout[n,k]  (f32 result)
// ---------------------------------------------------------------------------
__global__ __launch_bounds__(256)
void gemm_out(const _Float16* __restrict__ Mh, const _Float16* __restrict__ W,
              float* __restrict__ Out) {
    int lane = threadIdx.x & 31;
    int wave = threadIdx.x >> 5;
    int rowBase = blockIdx.x * 128 + wave * 16;
    int colBase = blockIdx.y * 64;

    v8f acc[4] = {};
    for (int k0 = 0; k0 < D_MODEL; k0 += 32) {
        v16h a = load_a_frag(Mh, D_MODEL, rowBase, k0, lane);
        __builtin_prefetch(Mh + (size_t)(rowBase + (lane & 15)) * D_MODEL + k0 + 64, 0, 1);
#pragma unroll
        for (int nb = 0; nb < 4; ++nb) {
            v16h b = load_bt_frag(W, D_MODEL, colBase + nb * 16, k0, lane);
            acc[nb] = __builtin_amdgcn_wmma_f32_16x16x32_f16(
                false, a, false, b, (short)0, acc[nb], false, false);
        }
    }
    int rowOff = (lane >> 4) * 8;
#pragma unroll
    for (int nb = 0; nb < 4; ++nb) {
        int n = colBase + nb * 16 + (lane & 15);
#pragma unroll
        for (int r = 0; r < 8; ++r) {
            int m = rowBase + r + rowOff;
            Out[(size_t)m * D_MODEL + n] = acc[nb][r];
        }
    }
}

// ---------------------------------------------------------------------------
// Launch
// ---------------------------------------------------------------------------
extern "C" void kernel_launch(void* const* d_in, const int* in_sizes, int n_in,
                              void* d_out, int out_size, void* d_ws, size_t ws_size,
                              hipStream_t stream) {
    const float* x     = (const float*)d_in[0];
    const float* W_qkv = (const float*)d_in[1];
    const float* W_out = (const float*)d_in[2];
    float*       out   = (float*)d_out;

    char* ws = (char*)d_ws;
    size_t off = 0;
    auto alloc = [&](size_t bytes) -> void* {
        void* p = ws + off;
        off += (bytes + 255) & ~(size_t)255;
        return p;
    };

    const size_t hB = sizeof(_Float16);
    _Float16* Xh    = (_Float16*)alloc((size_t)M_ROWS * D_MODEL * hB);
    _Float16* Wqkvh = (_Float16*)alloc((size_t)QKV_N * D_MODEL * hB);
    _Float16* Wouth = (_Float16*)alloc((size_t)D_MODEL * D_MODEL * hB);
    _Float16* Qh    = (_Float16*)alloc((size_t)BATCH * N_HEADS * SEQ * D_HEAD * hB);
    _Float16* Kh    = (_Float16*)alloc((size_t)BATCH * N_HEADS * SEQ * D_HEAD * hB);
    _Float16* Vh    = (_Float16*)alloc((size_t)BATCH * N_HEADS * SEQ * D_HEAD * hB);
    _Float16* AVh   = (_Float16*)alloc((size_t)BATCH * N_HEADS * SEQ * D_HEAD * hB);
    _Float16* AAVh  = (_Float16*)alloc((size_t)BATCH * N_HEADS * SEQ * D_HEAD * hB);
    _Float16* Mh    = (_Float16*)alloc((size_t)M_ROWS * D_MODEL * hB);

    int nX = M_ROWS * D_MODEL, nWq = QKV_N * D_MODEL, nWo = D_MODEL * D_MODEL;
    cvt_f32_to_f16<<<(nX  + 255) / 256, 256, 0, stream>>>(x,     Xh,    nX);
    cvt_f32_to_f16<<<(nWq + 255) / 256, 256, 0, stream>>>(W_qkv, Wqkvh, nWq);
    cvt_f32_to_f16<<<(nWo + 255) / 256, 256, 0, stream>>>(W_out, Wouth, nWo);

    gemm_qkv<<<dim3(M_ROWS / 128, QKV_N / 64), 256, 0, stream>>>(
        Xh, Wqkvh, Qh, Kh, Vh);

    dim3 agrid(SEQ / 64, BATCH * N_HEADS);
    flash_attn<<<agrid, 128, 0, stream>>>(Qh, Kh, Vh,  AVh);   // AV  = A @ V
    flash_attn<<<agrid, 128, 0, stream>>>(Qh, Kh, AVh, AAVh);  // AAV = A @ AV

    int nM = BATCH * SEQ * D_MODEL;
    combine_av<<<(nM + 255) / 256, 256, 0, stream>>>(AVh, AAVh, Mh);

    gemm_out<<<dim3(M_ROWS / 128, D_MODEL / 64), 256, 0, stream>>>(Mh, Wouth, out);
}
